// CDUserItemAttention_30047591203213
// MI455X (gfx1250) — matrix operand order repaired and checked by reading, not compiled, not run on hardware
//
#include <hip/hip_runtime.h>

// ---------------------------------------------------------------------------
// Types for CDNA5 WMMA (wave32): v_wmma_f32_16x16x32_f16
// ---------------------------------------------------------------------------
typedef __attribute__((ext_vector_type(16))) _Float16 v16h;
typedef __attribute__((ext_vector_type(4)))  _Float16 v4h;
typedef __attribute__((ext_vector_type(8)))  float    v8f;

union FragH {
  v16h  v;
  uint4 q[2];
};

#define NROW 4096
#define DDIM 512
#define ACTE ((size_t)NROW * DDIM)   // 2,097,152 elements
#define WELE ((size_t)DDIM * DDIM)   //   262,144 elements
#define ALPHA_F 0.9f
#define INV_SQRT_D 0.04419417382415922f  // 1/sqrt(512)

// ---------------------------------------------------------------------------
// Elementwise f32 -> f16 conversion (activations), 4 elements / thread
// ---------------------------------------------------------------------------
__global__ void cd_conv_act(const float* __restrict__ src,
                            _Float16* __restrict__ dst) {
  size_t i = ((size_t)blockIdx.x * blockDim.x + threadIdx.x) * 4;
  float4 f = *(const float4*)(src + i);
  v4h h;
  h.x = (_Float16)f.x; h.y = (_Float16)f.y;
  h.z = (_Float16)f.z; h.w = (_Float16)f.w;
  *(v4h*)(dst + i) = h;
}

// Weight convert + transpose: Wt[n*512 + k] = W[k*512 + n]
__global__ void cd_conv_wT(const float* __restrict__ W,
                           _Float16* __restrict__ Wt) {
  int idx = blockIdx.x * blockDim.x + threadIdx.x;  // 0 .. 512*512-1
  int k = idx & (DDIM - 1);
  int n = idx >> 9;
  Wt[idx] = (_Float16)W[(size_t)k * DDIM + n];
}

// ---------------------------------------------------------------------------
// One wave computes a 32(M) x 64(N) f32 tile: each B fragment is loaded once
// and fed to two WMMAs (two A row-blocks) -> halves B traffic per FLOP.
// Second operand stored n-major: Bn[n*ldb + k].
// acc[i*4+j] covers rows m0+i*16.. and cols n0+j*16..
// ---------------------------------------------------------------------------
__device__ __forceinline__ void wmma_tile_32x64(
    const _Float16* __restrict__ A, int lda,
    const _Float16* __restrict__ Bn, int ldb,
    int m0, int n0, int Kdim, v8f acc[8]) {
  const int lane = threadIdx.x & 31;
  const int lm   = lane & 15;
  const int half = lane >> 4;
  const _Float16* arow0 = A + (size_t)(m0 + lm) * lda;
  const _Float16* arow1 = arow0 + (size_t)16 * lda;
  for (int k0 = 0; k0 < Kdim; k0 += 32) {
    FragH a0, a1;
    // A lane layout: K = [k0+half*8, +8) and [k0+16+half*8, +8)
    a0.q[0] = *(const uint4*)(arow0 + k0 + half * 8);
    a0.q[1] = *(const uint4*)(arow0 + k0 + half * 8 + 16);
    a1.q[0] = *(const uint4*)(arow1 + k0 + half * 8);
    a1.q[1] = *(const uint4*)(arow1 + k0 + half * 8 + 16);
#pragma unroll
    for (int j = 0; j < 4; ++j) {
      // B lane layout: column n fixed, K = [k0+half*8, +8) and [k0+16+half*8, +8)
      const _Float16* brow =
          Bn + (size_t)(n0 + j * 16 + lm) * ldb + k0 + half * 8;
      FragH b;
      b.q[0] = *(const uint4*)(brow);
      b.q[1] = *(const uint4*)(brow + 16);
      acc[j] = __builtin_amdgcn_wmma_f32_16x16x32_f16(
          false, a0.v, false, b.v, (short)0, acc[j], false, false);
      acc[4 + j] = __builtin_amdgcn_wmma_f32_16x16x32_f16(
          false, a1.v, false, b.v, (short)0, acc[4 + j], false, false);
    }
  }
}

// ---------------------------------------------------------------------------
// Projections: P[p] = X[p] @ W[p] + b[p]   (W pre-transposed in ws)
// p = 0..4 : q_u, k_u, v_u, q_i, k_i.  v_u additionally stored transposed.
// grid (128, 8, 5), block 32
// ---------------------------------------------------------------------------
__global__ void cd_proj_gemm(const _Float16* __restrict__ actH,
                             const _Float16* __restrict__ wtH,
                             const float* __restrict__ b0,
                             const float* __restrict__ b1,
                             const float* __restrict__ b2,
                             const float* __restrict__ b3,
                             const float* __restrict__ b4,
                             _Float16* __restrict__ projH,
                             _Float16* __restrict__ vuT) {
  const int p = blockIdx.z;
  const _Float16* A  = actH + (size_t)p * ACTE;
  const _Float16* Bn = wtH  + (size_t)p * WELE;
  const float* bias = (p == 0) ? b0 : (p == 1) ? b1 : (p == 2) ? b2
                    : (p == 3) ? b3 : b4;
  _Float16* D = projH + (size_t)p * ACTE;

  const int m0 = blockIdx.x * 32;
  const int n0 = blockIdx.y * 64;
  const int lane = threadIdx.x & 31;
  const int lm = lane & 15, half = lane >> 4;

  v8f acc[8] = {};
  wmma_tile_32x64(A, DDIM, Bn, DDIM, m0, n0, DDIM, acc);

#pragma unroll
  for (int j = 0; j < 4; ++j) {
    int n = n0 + j * 16 + lm;
    float bv = bias[n];
#pragma unroll
    for (int i = 0; i < 2; ++i) {
#pragma unroll
      for (int r = 0; r < 8; ++r) {
        int m = m0 + i * 16 + r + 8 * half;
        _Float16 h = (_Float16)(acc[i * 4 + j][r] + bv);
        D[(size_t)m * DDIM + n] = h;
        if (p == 2) vuT[(size_t)n * NROW + m] = h;  // v_u^T [512,4096]
      }
    }
  }
}

// ---------------------------------------------------------------------------
// Scores: S = (Q @ K^T) / sqrt(d).  z=0 -> user (d_out attn region),
// z=1 -> item (ws).  grid (128, 64, 2), block 32
// ---------------------------------------------------------------------------
__global__ void cd_score_gemm(const _Float16* __restrict__ projH,
                              float* __restrict__ Su,
                              float* __restrict__ Si) {
  const int which = blockIdx.z;
  const _Float16* Q  = projH + (size_t)(which == 0 ? 0 : 3) * ACTE;
  const _Float16* Km = projH + (size_t)(which == 0 ? 1 : 4) * ACTE;
  float* S = (which == 0) ? Su : Si;

  const int m0 = blockIdx.x * 32;
  const int n0 = blockIdx.y * 64;
  const int lane = threadIdx.x & 31;
  const int lm = lane & 15, half = lane >> 4;

  v8f acc[8] = {};
  wmma_tile_32x64(Q, DDIM, Km, DDIM, m0, n0, DDIM, acc);

#pragma unroll
  for (int j = 0; j < 4; ++j) {
    int n = n0 + j * 16 + lm;
#pragma unroll
    for (int i = 0; i < 2; ++i) {
#pragma unroll
      for (int r = 0; r < 8; ++r) {
        int m = m0 + i * 16 + r + 8 * half;
        S[(size_t)m * NROW + n] = acc[i * 4 + j][r] * INV_SQRT_D;
      }
    }
  }
}

// ---------------------------------------------------------------------------
// Per-row softmax of Su and Si, blended: attn = a*sm(Su) + (1-a)*sm(Si).
// f32 attn written in place over Su (non-temporal: pure output), f16 copy
// kept regular (re-read by out GEMM).  grid 4096, block 256.
// ---------------------------------------------------------------------------
__device__ __forceinline__ float blk_reduce_max(float v, float* red) {
  int t = threadIdx.x;
  red[t] = v; __syncthreads();
  for (int s = 128; s > 0; s >>= 1) {
    if (t < s) red[t] = fmaxf(red[t], red[t + s]);
    __syncthreads();
  }
  float r = red[0]; __syncthreads();
  return r;
}
__device__ __forceinline__ float blk_reduce_sum(float v, float* red) {
  int t = threadIdx.x;
  red[t] = v; __syncthreads();
  for (int s = 128; s > 0; s >>= 1) {
    if (t < s) red[t] += red[t + s];
    __syncthreads();
  }
  float r = red[0]; __syncthreads();
  return r;
}

__global__ void cd_softmax_blend(float* __restrict__ Su,
                                 const float* __restrict__ Si,
                                 _Float16* __restrict__ attnH) {
  __shared__ float red[256];
  const int row = blockIdx.x;
  const int t = threadIdx.x;
  float* ru = Su + (size_t)row * NROW;
  const float* ri = Si + (size_t)row * NROW;

  float su[16], si[16];
  float mu = -3.4e38f, mi = -3.4e38f;
#pragma unroll
  for (int i = 0; i < 16; ++i) {
    su[i] = ru[t + 256 * i];
    si[i] = __builtin_nontemporal_load(ri + t + 256 * i);  // Si dead after
    mu = fmaxf(mu, su[i]);
    mi = fmaxf(mi, si[i]);
  }
  mu = blk_reduce_max(mu, red);
  mi = blk_reduce_max(mi, red);

  float lu = 0.f, li = 0.f;
#pragma unroll
  for (int i = 0; i < 16; ++i) {
    su[i] = __expf(su[i] - mu); lu += su[i];
    si[i] = __expf(si[i] - mi); li += si[i];
  }
  lu = blk_reduce_sum(lu, red);
  li = blk_reduce_sum(li, red);

  const float cu = ALPHA_F / lu;
  const float ci = (1.0f - ALPHA_F) / li;
  _Float16* ah = attnH + (size_t)row * NROW;
#pragma unroll
  for (int i = 0; i < 16; ++i) {
    float a = su[i] * cu + si[i] * ci;
    __builtin_nontemporal_store(a, ru + t + 256 * i);  // final attn f32 output
    ah[t + 256 * i] = (_Float16)a;                      // f16 copy, re-read
  }
}

// ---------------------------------------------------------------------------
// out = attn_f16 @ v_u : [4096,4096]x[4096,512] -> f32.  B is vuT (n-major).
// grid (128, 8), block 32
// ---------------------------------------------------------------------------
__global__ void cd_out_gemm(const _Float16* __restrict__ attnH,
                            const _Float16* __restrict__ vuT,
                            float* __restrict__ out) {
  const int m0 = blockIdx.x * 32;
  const int n0 = blockIdx.y * 64;
  const int lane = threadIdx.x & 31;
  const int lm = lane & 15, half = lane >> 4;

  v8f acc[8] = {};
  wmma_tile_32x64(attnH, NROW, vuT, NROW, m0, n0, NROW, acc);

#pragma unroll
  for (int j = 0; j < 4; ++j) {
    int n = n0 + j * 16 + lm;
#pragma unroll
    for (int i = 0; i < 2; ++i) {
#pragma unroll
      for (int r = 0; r < 8; ++r) {
        int m = m0 + i * 16 + r + 8 * half;
        __builtin_nontemporal_store(acc[i * 4 + j][r],
                                    out + (size_t)m * DDIM + n);
      }
    }
  }
}

// ---------------------------------------------------------------------------
// Host-side orchestration
// ---------------------------------------------------------------------------
extern "C" void kernel_launch(void* const* d_in, const int* in_sizes, int n_in,
                              void* d_out, int out_size, void* d_ws, size_t ws_size,
                              hipStream_t stream) {
  (void)in_sizes; (void)n_in; (void)out_size; (void)ws_size;

  // Inputs (setup_inputs order):
  // 0..4  : Q_user, K_user, V_user, Q_item, K_item        [4096,512] f32
  // 5+2p  : W_{q_u,k_u,v_u,q_i,k_i}[p]                    [512,512]  f32
  // 6+2p  : b_{...}[p]                                    [512]      f32
  const float* bias[5];
  for (int p = 0; p < 5; ++p) bias[p] = (const float*)d_in[6 + 2 * p];

  // Workspace layout (f16 elements unless noted):
  _Float16* actH  = (_Float16*)d_ws;            // 5 * ACTE
  _Float16* wtH   = actH  + 5 * ACTE;           // 5 * WELE (transposed weights)
  _Float16* projH = wtH   + 5 * WELE;           // 5 * ACTE (q_u,k_u,v_u,q_i,k_i)
  _Float16* vuT   = projH + 5 * ACTE;           // ACTE     (v_u transposed)
  float*    Si    = (float*)(vuT + ACTE);       // 4096*4096 f32 (item scores)
  _Float16* attnH = (_Float16*)(Si + (size_t)NROW * NROW);  // 4096*4096 f16

  // d_out layout: [0, 2M) = out f32; [2M, 18M) = attn f32 (also holds Su).
  float* outF = (float*)d_out;
  float* Su   = outF + ACTE;

  // 1) Convert activations + weights(T) to f16
  for (int p = 0; p < 5; ++p) {
    cd_conv_act<<<(int)(ACTE / 1024), 256, 0, stream>>>(
        (const float*)d_in[p], actH + (size_t)p * ACTE);
    cd_conv_wT<<<(int)(WELE / 256), 256, 0, stream>>>(
        (const float*)d_in[5 + 2 * p], wtH + (size_t)p * WELE);
  }

  // 2) Projections (WMMA)
  cd_proj_gemm<<<dim3(NROW / 32, DDIM / 64, 5), 32, 0, stream>>>(
      actH, wtH, bias[0], bias[1], bias[2], bias[3], bias[4], projH, vuT);

  // 3) Score matrices (WMMA): Su -> d_out attn region, Si -> ws
  cd_score_gemm<<<dim3(NROW / 32, NROW / 64, 2), 32, 0, stream>>>(
      projH, Su, Si);

  // 4) Row softmax + blend (attn f32 in place, f16 copy to ws)
  cd_softmax_blend<<<NROW, 256, 0, stream>>>(Su, Si, attnH);

  // 5) out = attn @ v_u (WMMA)
  cd_out_gemm<<<dim3(NROW / 32, DDIM / 64), 32, 0, stream>>>(
      attnH, vuT, outF);
}
